// HierarchicalSparseAttention_11149735100726
// MI455X (gfx1250) — compile-verified, matrix operand order checked by reading, not run on hardware
//
#include <hip/hip_runtime.h>

// ---------------------------------------------------------------------------
// HierarchicalSparseAttention for MI455X (gfx1250), wave32 + WMMA bf16 path,
// async-to-LDS double-buffered GEMM staging (ASYNCcnt pipeline).
// B=2, L=4096, D=1024, H=16, DH=64.
// ---------------------------------------------------------------------------

typedef __attribute__((ext_vector_type(16))) __bf16 v16bf;
typedef __attribute__((ext_vector_type(8)))  float  v8f;

#define BATCH 2
#define SEQ   4096
#define DMODEL 1024
#define NHEAD 16
#define DHEAD 64
#define LEVELS 12
#define FLATN 8191            // 4096+2048+...+1
#define NTOK  (BATCH*SEQ)     // 8192 rows in the GEMMs

__device__ __forceinline__ unsigned short f2bf(float f) {
    unsigned int u = __float_as_uint(f);
    unsigned int r = (u + 0x7FFFu + ((u >> 16) & 1u)) >> 16;   // RNE
    return (unsigned short)r;
}

// ---------------------------------------------------------------------------
// fp32 -> bf16 cast
// ---------------------------------------------------------------------------
__global__ void cast_f32_to_bf16(const float* __restrict__ in,
                                 unsigned short* __restrict__ out, int n) {
    int i = blockIdx.x * blockDim.x + threadIdx.x;
    if (i < n) out[i] = f2bf(in[i]);
}

// ---------------------------------------------------------------------------
// Async global->LDS copy of 32 contiguous bytes (2 x b128).
// Per ISA §10: INST_OFFSET is added to BOTH the LDS and global address,
// so a single VGPR pair + LDS offset covers both halves. Tracked by ASYNCcnt.
// ---------------------------------------------------------------------------
__device__ __forceinline__ void async_copy32(unsigned lds_off,
                                             const unsigned short* g) {
    asm volatile("global_load_async_to_lds_b128 %0, %1, off\n\t"
                 "global_load_async_to_lds_b128 %0, %1, off offset:16"
                 :: "v"(lds_off), "v"(g) : "memory");
}
#define ASYNC_WAIT4() asm volatile("s_wait_asynccnt 4" ::: "memory")
#define ASYNC_WAIT0() asm volatile("s_wait_asynccnt 0" ::: "memory")

// ---------------------------------------------------------------------------
// GEMM: C[M,N] = A[M,K] @ W[N,K]^T (+ bias), A/W bf16, C fp32.
// 128x128x32 tile, 256 threads = 8 waves, v_wmma_f32_16x16x32_bf16.
// Double-buffered LDS fed by async-to-LDS loads; requires K % 64 == 0.
// ---------------------------------------------------------------------------
#define BM 128
#define BN 128
#define BK 32
#define LDSTR 48   // padded LDS row stride (elements): 96 bytes, 32B aligned

__global__ __launch_bounds__(256)
void gemm_bf16_wmma(const unsigned short* __restrict__ A,
                    const unsigned short* __restrict__ W,
                    float* __restrict__ C,
                    const float* __restrict__ bias,
                    int M, int N, int K, int addBias) {
    __shared__ unsigned short As[2][BM * LDSTR];
    __shared__ unsigned short Bs[2][BN * LDSTR];

    const int tid  = threadIdx.x;
    const int wave = tid >> 5;
    const int lane = tid & 31;
    const int m0 = blockIdx.y * BM;
    const int n0 = blockIdx.x * BN;

    // wave tile: 4 waves along M (32 rows each), 2 waves along N (64 cols each)
    const int wm = (wave >> 1) * 32;
    const int wn = (wave & 1) * 64;

    v8f acc[2][4] = {};

    // async staging map: thread t covers 32 B of row (t>>1), half (t&1)
    const int r    = tid >> 1;           // 0..127
    const int half = tid & 1;            // 0 or 1 (16-element halves)
    const unsigned short* gA = A + (size_t)(m0 + r) * K + half * 16;
    const unsigned short* gB = W + (size_t)(n0 + r) * K + half * 16;
    // LDS byte offsets (generic address truncates to LDS offset per aperture map)
    const unsigned ldsA0 = (unsigned)(size_t)&As[0][r * LDSTR + half * 16];
    const unsigned ldsA1 = (unsigned)(size_t)&As[1][r * LDSTR + half * 16];
    const unsigned ldsB0 = (unsigned)(size_t)&Bs[0][r * LDSTR + half * 16];
    const unsigned ldsB1 = (unsigned)(size_t)&Bs[1][r * LDSTR + half * 16];

    const int lrow  = lane & 15;         // fragment row within 16x16 tile
    const int lkoff = (lane >> 4) * 16;  // K half per lane group

#define ISSUE(la, lb, kk) do { async_copy32((la), gA + (kk)); \
                               async_copy32((lb), gB + (kk)); } while (0)
#define COMPUTE(bufi) do {                                                    \
        v16bf af_[2], bf_[4];                                                 \
        _Pragma("unroll")                                                     \
        for (int i = 0; i < 2; ++i)                                           \
            af_[i] = *(const v16bf*)&As[bufi][(wm + i*16 + lrow)*LDSTR + lkoff]; \
        _Pragma("unroll")                                                     \
        for (int j = 0; j < 4; ++j)                                           \
            bf_[j] = *(const v16bf*)&Bs[bufi][(wn + j*16 + lrow)*LDSTR + lkoff]; \
        _Pragma("unroll")                                                     \
        for (int i = 0; i < 2; ++i)                                           \
            _Pragma("unroll")                                                 \
            for (int j = 0; j < 4; ++j)                                       \
                acc[i][j] = __builtin_amdgcn_wmma_f32_16x16x32_bf16(          \
                    false, af_[i], false, bf_[j], (short)0, acc[i][j],        \
                    false, false);                                            \
    } while (0)

    // prologue: stage k=0 into buffer 0 (4 async ops per wave outstanding)
    ISSUE(ldsA0, ldsB0, 0);

    int k0 = 0;
    for (; k0 + 2 * BK < K; k0 += 2 * BK) {
        ISSUE(ldsA1, ldsB1, k0 + BK);    // prefetch next tile into buf1
        ASYNC_WAIT4();                   // older 4 (buf0) retired, in-order
        __syncthreads();
        COMPUTE(0);
        __syncthreads();                 // all waves done reading buf0
        ISSUE(ldsA0, ldsB0, k0 + 2 * BK);
        ASYNC_WAIT4();                   // buf1 loads retired
        __syncthreads();
        COMPUTE(1);
        __syncthreads();
    }
    // tail: exactly two k-steps remain (K multiple of 64)
    ISSUE(ldsA1, ldsB1, k0 + BK);
    ASYNC_WAIT4();
    __syncthreads();
    COMPUTE(0);
    __syncthreads();
    ASYNC_WAIT0();
    __syncthreads();
    COMPUTE(1);
#undef ISSUE
#undef COMPUTE

    // C layout per ISA: VGPR r -> lanes 0-15: M=r, N=lane; lanes 16-31: M=r+8
    #pragma unroll
    for (int i = 0; i < 2; ++i) {
        #pragma unroll
        for (int j = 0; j < 4; ++j) {
            const int row0 = m0 + wm + i * 16 + (lane >> 4) * 8;
            const int col  = n0 + wn + j * 16 + (lane & 15);
            float bv = addBias ? bias[col] : 0.0f;
            #pragma unroll
            for (int rr = 0; rr < 8; ++rr)
                C[(size_t)(row0 + rr) * N + col] = acc[i][j][rr] + bv;
        }
    }
}

// ---------------------------------------------------------------------------
// Pyramid level 0: transpose (B,L,H*DH) -> (B,H,node0..L-1,DH)
// ---------------------------------------------------------------------------
__global__ void pyramid_level0(const float* __restrict__ Kf,
                               const float* __restrict__ Vf,
                               float* __restrict__ flatK,
                               float* __restrict__ flatV) {
    size_t i = (size_t)blockIdx.x * blockDim.x + threadIdx.x; // over B*L*D
    int d = i & (DHEAD - 1);
    size_t t = i >> 6;
    int h = t & (NHEAD - 1); t >>= 4;
    int l = t & (SEQ - 1);
    int b = (int)(t >> 12);
    size_t dst = (((size_t)(b * NHEAD + h)) * FLATN + l) * DHEAD + d;
    flatK[dst] = Kf[i];
    flatV[dst] = Vf[i];
}

// ---------------------------------------------------------------------------
// Pyramid reduce: level j -> j+1. K averages pairs, V sums pairs.
// ---------------------------------------------------------------------------
__global__ void pyramid_reduce(float* __restrict__ flatK,
                               float* __restrict__ flatV,
                               int srcOff, int dstOff, int halfLc) {
    size_t i = (size_t)blockIdx.x * blockDim.x + threadIdx.x;
    size_t total = (size_t)BATCH * NHEAD * halfLc * DHEAD;
    if (i >= total) return;
    int d = i & (DHEAD - 1);
    size_t t = i >> 6;
    int node = (int)(t % halfLc);
    int bh   = (int)(t / halfLc);
    size_t base = (size_t)bh * FLATN * DHEAD;
    size_t s0 = base + (size_t)(srcOff + 2 * node) * DHEAD + d;
    size_t s1 = s0 + DHEAD;
    size_t dd = base + (size_t)(dstOff + node) * DHEAD + d;
    flatK[dd] = 0.5f * (flatK[s0] + flatK[s1]);
    flatV[dd] = flatV[s0] + flatV[s1];
}

// ---------------------------------------------------------------------------
// Sparse attention: one wave32 per (b,h,l). Lane owns dims {lane, lane+32}.
// 13 logits (self + 12 gathered pyramid neighbors), masked softmax,
// weighted V sum. Output stored bf16 in (B,L,H*DH) layout for the Wo GEMM.
// ---------------------------------------------------------------------------
__global__ __launch_bounds__(256)
void sparse_attn(const float* __restrict__ Qf,
                 const float* __restrict__ flatK,
                 const float* __restrict__ flatV,
                 const int* __restrict__ idxTab,
                 const unsigned char* __restrict__ cmask,
                 unsigned short* __restrict__ outbf) {
    const int wid  = blockIdx.x * (blockDim.x >> 5) + (threadIdx.x >> 5);
    const int lane = threadIdx.x & 31;
    const int l = wid & (SEQ - 1);
    const int h = (wid >> 12) & (NHEAD - 1);
    const int b = wid >> 16;
    const float scale = 0.125f;  // 1/sqrt(64)

    const float* q = Qf + (((size_t)b * SEQ + l) * DMODEL + h * DHEAD);
    const float q0 = q[lane];
    const float q1 = q[lane + 32];

    const size_t hb = (size_t)(b * NHEAD + h) * FLATN * DHEAD;
    const float* kb = flatK + hb;
    const float* vb = flatV + hb;

    float logits[LEVELS + 1];
    int   nodes[LEVELS + 1];

    nodes[0] = l;  // self-logit uses level-0 K = K itself
    {
        const float* kp = kb + (size_t)l * DHEAD;
        float p = q0 * kp[lane] + q1 * kp[lane + 32];
        #pragma unroll
        for (int o = 16; o > 0; o >>= 1) p += __shfl_xor(p, o, 32);
        logits[0] = p * scale;
    }
    #pragma unroll
    for (int lvl = 0; lvl < LEVELS; ++lvl) {
        int node = idxTab[l * LEVELS + lvl];
        if (node < 0) node += FLATN;      // jnp negative-index wrap semantics
        nodes[lvl + 1] = node;
        const float* kp = kb + (size_t)node * DHEAD;
        float p = q0 * kp[lane] + q1 * kp[lane + 32];
        #pragma unroll
        for (int o = 16; o > 0; o >>= 1) p += __shfl_xor(p, o, 32);
        p *= scale;
        if (cmask[l * LEVELS + lvl]) p = __int_as_float(0xff800000); // -inf
        logits[lvl + 1] = p;
    }

    float mx = logits[0];
    #pragma unroll
    for (int n = 1; n <= LEVELS; ++n) mx = fmaxf(mx, logits[n]);
    float w[LEVELS + 1], se = 0.0f;
    #pragma unroll
    for (int n = 0; n <= LEVELS; ++n) { w[n] = __expf(logits[n] - mx); se += w[n]; }
    const float inv = 1.0f / se;

    float o0 = 0.0f, o1 = 0.0f;
    #pragma unroll
    for (int n = 0; n <= LEVELS; ++n) {
        const float* vp = vb + (size_t)nodes[n] * DHEAD;
        o0 += w[n] * vp[lane];
        o1 += w[n] * vp[lane + 32];
    }
    o0 *= inv; o1 *= inv;

    const size_t ob = ((size_t)b * SEQ + l) * DMODEL + h * DHEAD;
    outbf[ob + lane]      = f2bf(o0);
    outbf[ob + lane + 32] = f2bf(o1);
}

// ---------------------------------------------------------------------------
// Host-side orchestration (graph-capture safe: all launches on `stream`).
// ---------------------------------------------------------------------------
static inline size_t align256(size_t x) { return (x + 255) & ~(size_t)255; }

extern "C" void kernel_launch(void* const* d_in, const int* in_sizes, int n_in,
                              void* d_out, int out_size, void* d_ws, size_t ws_size,
                              hipStream_t stream) {
    const float* query = (const float*)d_in[0];
    const float* key   = (const float*)d_in[1];
    const float* value = (const float*)d_in[2];
    const float* Wq    = (const float*)d_in[3];
    const float* Wk    = (const float*)d_in[4];
    const float* Wv    = (const float*)d_in[5];
    const float* Wo    = (const float*)d_in[6];
    const float* bo    = (const float*)d_in[7];
    const int*   idxT  = (const int*)d_in[8];
    const unsigned char* cmask = (const unsigned char*)d_in[9];
    float* out = (float*)d_out;

    const size_t nTok   = (size_t)NTOK * DMODEL;   // 8192*1024
    const size_t nW     = (size_t)DMODEL * DMODEL; // 1024*1024
    const size_t nFlat  = (size_t)BATCH * NHEAD * FLATN * DHEAD;

    char* ws = (char*)d_ws;
    size_t off = 0;
    unsigned short* qbf   = (unsigned short*)(ws + off); off = align256(off + nTok * 2);
    unsigned short* kbf   = (unsigned short*)(ws + off); off = align256(off + nTok * 2);
    unsigned short* vbf   = (unsigned short*)(ws + off); off = align256(off + nTok * 2);
    unsigned short* wqbf  = (unsigned short*)(ws + off); off = align256(off + nW * 2);
    unsigned short* wkbf  = (unsigned short*)(ws + off); off = align256(off + nW * 2);
    unsigned short* wvbf  = (unsigned short*)(ws + off); off = align256(off + nW * 2);
    unsigned short* wobf  = (unsigned short*)(ws + off); off = align256(off + nW * 2);
    float*          Qf    = (float*)(ws + off);          off = align256(off + nTok * 4);
    float*          Kf    = (float*)(ws + off);          off = align256(off + nTok * 4);
    float*          Vf    = (float*)(ws + off);          off = align256(off + nTok * 4);
    float*          flatK = (float*)(ws + off);          off = align256(off + nFlat * 4);
    float*          flatV = (float*)(ws + off);          off = align256(off + nFlat * 4);
    unsigned short* attnbf = (unsigned short*)(ws + off); off = align256(off + nTok * 2);
    (void)ws_size; (void)in_sizes; (void)n_in; (void)out_size;

    // 1) casts to bf16
    {
        int n = (int)nTok, g = (n + 255) / 256;
        cast_f32_to_bf16<<<g, 256, 0, stream>>>(query, qbf, n);
        cast_f32_to_bf16<<<g, 256, 0, stream>>>(key,   kbf, n);
        cast_f32_to_bf16<<<g, 256, 0, stream>>>(value, vbf, n);
        int nw = (int)nW, gw = (nw + 255) / 256;
        cast_f32_to_bf16<<<gw, 256, 0, stream>>>(Wq, wqbf, nw);
        cast_f32_to_bf16<<<gw, 256, 0, stream>>>(Wk, wkbf, nw);
        cast_f32_to_bf16<<<gw, 256, 0, stream>>>(Wv, wvbf, nw);
        cast_f32_to_bf16<<<gw, 256, 0, stream>>>(Wo, wobf, nw);
    }

    // 2) Q/K/V projections: (8192,1024) @ (1024,1024)^T
    dim3 ggrid(DMODEL / BN, NTOK / BM);
    gemm_bf16_wmma<<<ggrid, 256, 0, stream>>>(qbf, wqbf, Qf, nullptr,
                                              NTOK, DMODEL, DMODEL, 0);
    gemm_bf16_wmma<<<ggrid, 256, 0, stream>>>(kbf, wkbf, Kf, nullptr,
                                              NTOK, DMODEL, DMODEL, 0);
    gemm_bf16_wmma<<<ggrid, 256, 0, stream>>>(vbf, wvbf, Vf, nullptr,
                                              NTOK, DMODEL, DMODEL, 0);

    // 3) pyramid
    {
        int g = (int)(nTok / 256);
        pyramid_level0<<<g, 256, 0, stream>>>(Kf, Vf, flatK, flatV);
        int srcOff = 0, Lc = SEQ;
        for (int j = 0; j < LEVELS; ++j) {
            int half = Lc >> 1;
            int dstOff = srcOff + Lc;
            size_t total = (size_t)BATCH * NHEAD * half * DHEAD;
            int gr = (int)((total + 255) / 256);
            pyramid_reduce<<<gr, 256, 0, stream>>>(flatK, flatV, srcOff, dstOff, half);
            srcOff = dstOff;
            Lc = half;
        }
    }

    // 4) sparse attention: one wave per (b,h,l): 131072 waves / 8 per block
    {
        int g = (BATCH * NHEAD * SEQ) / 8;
        sparse_attn<<<g, 256, 0, stream>>>(Qf, flatK, flatV, idxT, cmask, attnbf);
    }

    // 5) output projection + bias -> d_out
    gemm_bf16_wmma<<<ggrid, 256, 0, stream>>>(attnbf, wobf, out, bo,
                                              NTOK, DMODEL, DMODEL, 1);
}
